// SAGEEncoder_16174846836858
// MI455X (gfx1250) — matrix operand order repaired and checked by reading, not compiled, not run on hardware
//
#include <hip/hip_runtime.h>

// ---------------------------------------------------------------------------
// GraphSAGE 2-layer encoder for MI455X (gfx1250, wave32, WMMA f16->f32).
//
// Pipeline:
//   deg[n]   = sum_e [dst==n]                         (f32 atomics, L2)
//   agg1     = scatter-add x[src] onto dst            (f32 atomics, L2)
//   h        = relu(agg1/deg @ W1_l.T + b1 + x @ W1_r.T)   (WMMA f16)
//   agg2     = scatter-add h[src] onto dst
//   out      = agg2/deg @ W2_l.T + b2 + h @ W2_r.T         (WMMA f16)
// ---------------------------------------------------------------------------

#define NNODES 50000
#define NEDGES 800000
#define INFEAT 128
#define HIDDEN 256
#define OUTF   64

typedef __attribute__((ext_vector_type(16))) _Float16 v16h;
typedef __attribute__((ext_vector_type(8)))  _Float16 v8h;
typedef __attribute__((ext_vector_type(8)))  float    v8f;

// ---------------------------- utility kernels ------------------------------

__global__ void zero_f32(float* __restrict__ p, long n) {
  long i = (long)blockIdx.x * blockDim.x + threadIdx.x;
  long stride = (long)gridDim.x * blockDim.x;
  for (; i < n; i += stride) p[i] = 0.0f;
}

__global__ void degree_count(const int* __restrict__ dst, float* __restrict__ cnt, int E) {
  int e = blockIdx.x * blockDim.x + threadIdx.x;
  if (e < E) atomicAdd(&cnt[dst[e]], 1.0f);
}

__global__ void inv_count(float* __restrict__ cnt, int n) {
  int i = blockIdx.x * blockDim.x + threadIdx.x;
  if (i < n) cnt[i] = 1.0f / fmaxf(cnt[i], 1.0f);
}

// one wave per edge; float4 gather (L2-resident), f32 atomic scatter
__global__ void scatter_add_edges(const float* __restrict__ X,
                                  const int* __restrict__ src,
                                  const int* __restrict__ dst,
                                  float* __restrict__ agg, int E, int F) {
  int wave = (blockIdx.x * blockDim.x + threadIdx.x) >> 5;
  int lane = threadIdx.x & 31;
  if (wave >= E) return;
  int s = src[wave];
  int d = dst[wave];
  const float4* xp = (const float4*)(X + (size_t)s * F);
  float* ap = agg + (size_t)d * F;
  int nvec = F >> 2;
  for (int c = lane; c < nvec; c += 32) {
    float4 v = xp[c];
    atomicAdd(ap + c * 4 + 0, v.x);
    atomicAdd(ap + c * 4 + 1, v.y);
    atomicAdd(ap + c * 4 + 2, v.z);
    atomicAdd(ap + c * 4 + 3, v.w);
  }
}

__global__ void f32_to_f16(const float* __restrict__ in, _Float16* __restrict__ out, long n) {
  long i = (long)blockIdx.x * blockDim.x + threadIdx.x;
  if (i < n) out[i] = (_Float16)in[i];
}

// out[i] = f16(in[i] * inv[i >> shift])   (per-row mean normalization)
__global__ void scale_rows_to_f16(const float* __restrict__ in, const float* __restrict__ inv,
                                  _Float16* __restrict__ out, long n, int shift) {
  long i = (long)blockIdx.x * blockDim.x + threadIdx.x;
  if (i < n) out[i] = (_Float16)(in[i] * inv[i >> shift]);
}

// ------------------------------- WMMA GEMM ---------------------------------
// D(MxN,f32) = A1(MxK) @ B1(NxK)^T + A2(MxK) @ B2(NxK)^T + bias(N), opt ReLU.
// One wave computes a 16x64 output strip (4 n-tiles; A fragment reused 4x).
// Fragment layouts used here (A and B share a contiguous k order, not the ISA table order):
//   A 16x32 f16: lane holds row m=lane&15; halfs [0..15]=K kb..kb+15,
//                kb = (lane>>4)*16.
//   B 32x16 f16: lane holds col n=lane&15; halfs [0..15]=K kb2..kb2+15,
//                kb2 = (lane>>4)*16.  With B = W^T this is 16 contiguous
//                halfs of W's row n -> direct 32-byte vector load.
//   C/D f32: vgpr r -> row mt*16 + (lane>>4)*8 + r, col n=lane&15.
template <int RELU>
__global__ void __launch_bounds__(256)
sage_gemm_f16(const _Float16* __restrict__ A1, const _Float16* __restrict__ B1,
              const _Float16* __restrict__ A2, const _Float16* __restrict__ B2,
              const float* __restrict__ bias, float* __restrict__ Df,
              _Float16* __restrict__ Dh, int M, int N, int K) {
  int tid  = blockIdx.x * blockDim.x + threadIdx.x;
  int wave = tid >> 5;
  int lane = threadIdx.x & 31;
  int ntn  = N >> 6;               // number of 64-wide column groups
  int mt   = wave / ntn;           // 16-row tile index (wave-uniform)
  int ng   = wave % ntn;
  if (mt * 16 >= M) return;        // wave-uniform: EXEC stays all-ones

  int lrow = lane & 15;
  int lhi  = lane >> 4;            // 0 or 1
  int kbA  = lhi * 16;
  int kbB  = lhi * 16;
  int m    = mt * 16 + lrow;

  v8f acc[4];
#pragma unroll
  for (int s = 0; s < 4; ++s)
#pragma unroll
    for (int r = 0; r < 8; ++r) acc[s][r] = 0.0f;

  const _Float16* a1row = A1 + (size_t)m * K;
  const _Float16* a2row = A2 + (size_t)m * K;

  for (int kk = 0; kk < K; kk += 32) {
    v16h a;
    {
      v8h lo = *(const v8h*)(a1row + kk + kbA);
      v8h hi = *(const v8h*)(a1row + kk + kbA + 8);
#pragma unroll
      for (int i = 0; i < 8; ++i) { a[i] = lo[i]; a[8 + i] = hi[i]; }
    }
#pragma unroll
    for (int s = 0; s < 4; ++s) {
      int n  = ng * 64 + s * 16 + lrow;
      v16h b = *(const v16h*)(B1 + (size_t)n * K + kk + kbB);
      acc[s] = __builtin_amdgcn_wmma_f32_16x16x32_f16(false, a, false, b,
                                                      (short)0, acc[s], false, false);
    }
    {
      v8h lo = *(const v8h*)(a2row + kk + kbA);
      v8h hi = *(const v8h*)(a2row + kk + kbA + 8);
#pragma unroll
      for (int i = 0; i < 8; ++i) { a[i] = lo[i]; a[8 + i] = hi[i]; }
    }
#pragma unroll
    for (int s = 0; s < 4; ++s) {
      int n  = ng * 64 + s * 16 + lrow;
      v16h b = *(const v16h*)(B2 + (size_t)n * K + kk + kbB);
      acc[s] = __builtin_amdgcn_wmma_f32_16x16x32_f16(false, a, false, b,
                                                      (short)0, acc[s], false, false);
    }
  }

#pragma unroll
  for (int s = 0; s < 4; ++s) {
    int n = ng * 64 + s * 16 + lrow;
    float bv = bias[n];
#pragma unroll
    for (int r = 0; r < 8; ++r) {
      float v = acc[s][r] + bv;
      if (RELU) v = fmaxf(v, 0.0f);
      int row = mt * 16 + lhi * 8 + r;
      Df[(size_t)row * N + n] = v;
      if (Dh) Dh[(size_t)row * N + n] = (_Float16)v;
    }
  }
}

// ------------------------------ launcher -----------------------------------

extern "C" void kernel_launch(void* const* d_in, const int* in_sizes, int n_in,
                              void* d_out, int out_size, void* d_ws, size_t ws_size,
                              hipStream_t stream) {
  const float* x   = (const float*)d_in[0];
  const int*   ei  = (const int*)d_in[1];   // [2, E]
  const float* W1l = (const float*)d_in[2];
  const float* W1r = (const float*)d_in[3];
  const float* b1  = (const float*)d_in[4];
  const float* W2l = (const float*)d_in[5];
  const float* W2r = (const float*)d_in[6];
  const float* b2  = (const float*)d_in[7];
  float* out = (float*)d_out;

  const int* src = ei;
  const int* dst = ei + NEDGES;

  // workspace layout (bytes), with region aliasing across phases
  const size_t OFF_AGG1  = 262144;                       // cnt occupies [0, 200000)
  const size_t SZ_AGG1   = (size_t)NNODES * INFEAT * 4;  // 25.6 MB
  const size_t OFF_AGG1H = OFF_AGG1 + SZ_AGG1;
  const size_t SZ_AGG1H  = (size_t)NNODES * INFEAT * 2;  // 12.8 MB
  const size_t OFF_XH    = OFF_AGG1H + SZ_AGG1H;
  const size_t SZ_XH     = SZ_AGG1H;                     // 12.8 MB
  const size_t OFF_H     = OFF_XH + SZ_XH;               // agg2 aliases [OFF_AGG1, OFF_H)
  const size_t SZ_H      = (size_t)NNODES * HIDDEN * 4;  // 51.2 MB
  const size_t OFF_HH    = OFF_H + SZ_H;
  const size_t SZ_HH     = (size_t)NNODES * HIDDEN * 2;  // 25.6 MB
  const size_t OFF_W     = OFF_HH + SZ_HH;

  char* ws = (char*)d_ws;
  float*     cnt   = (float*)(ws);
  float*     agg1  = (float*)(ws + OFF_AGG1);
  _Float16*  agg1h = (_Float16*)(ws + OFF_AGG1H);
  _Float16*  xh    = (_Float16*)(ws + OFF_XH);
  float*     agg2  = (float*)(ws + OFF_AGG1);            // alias (agg1/agg1h/xh dead)
  float*     hbuf  = (float*)(ws + OFF_H);
  _Float16*  agg2h = (_Float16*)(ws + OFF_H);            // alias (h f32 dead)
  _Float16*  hh    = (_Float16*)(ws + OFF_HH);
  _Float16*  w1lh  = (_Float16*)(ws + OFF_W);
  _Float16*  w1rh  = (_Float16*)(ws + OFF_W + 65536);
  _Float16*  w2lh  = (_Float16*)(ws + OFF_W + 131072);
  _Float16*  w2rh  = (_Float16*)(ws + OFF_W + 163840);

  const int T = 256;

  // ---- phase 1: degree + aggregate x ----
  {
    long nz = (long)(OFF_AGG1 + SZ_AGG1) / 4;            // cnt + agg1
    zero_f32<<<(int)((nz + T - 1) / T), T, 0, stream>>>((float*)ws, nz);
  }
  degree_count<<<(NEDGES + T - 1) / T, T, 0, stream>>>(dst, cnt, NEDGES);
  scatter_add_edges<<<(NEDGES * 32) / T, T, 0, stream>>>(x, src, dst, agg1, NEDGES, INFEAT);
  inv_count<<<(NNODES + T - 1) / T, T, 0, stream>>>(cnt, NNODES);

  // ---- f16 conversions ----
  {
    long n = (long)NNODES * INFEAT;
    f32_to_f16<<<(int)((n + T - 1) / T), T, 0, stream>>>(x, xh, n);
    f32_to_f16<<<(HIDDEN * INFEAT + T - 1) / T, T, 0, stream>>>(W1l, w1lh, HIDDEN * INFEAT);
    f32_to_f16<<<(HIDDEN * INFEAT + T - 1) / T, T, 0, stream>>>(W1r, w1rh, HIDDEN * INFEAT);
    f32_to_f16<<<(OUTF * HIDDEN + T - 1) / T, T, 0, stream>>>(W2l, w2lh, OUTF * HIDDEN);
    f32_to_f16<<<(OUTF * HIDDEN + T - 1) / T, T, 0, stream>>>(W2r, w2rh, OUTF * HIDDEN);
    scale_rows_to_f16<<<(int)((n + T - 1) / T), T, 0, stream>>>(agg1, cnt, agg1h, n, 7);
  }

  // ---- layer 1 GEMM: h = relu(agg1h@W1l^T + xh@W1r^T + b1), also emit f16 h
  {
    int waves = (NNODES / 16) * (HIDDEN / 64);           // 3125 * 4
    sage_gemm_f16<1><<<(waves * 32 + T - 1) / T, T, 0, stream>>>(
        agg1h, w1lh, xh, w1rh, b1, hbuf, hh, NNODES, HIDDEN, INFEAT);
  }

  // ---- phase 2: aggregate h ----
  {
    long nz = (long)SZ_H / 4;
    zero_f32<<<(int)((nz + T - 1) / T), T, 0, stream>>>(agg2, nz);
  }
  scatter_add_edges<<<(NEDGES * 32) / T, T, 0, stream>>>(hbuf, src, dst, agg2, NEDGES, HIDDEN);
  {
    long n = (long)NNODES * HIDDEN;
    scale_rows_to_f16<<<(int)((n + T - 1) / T), T, 0, stream>>>(agg2, cnt, agg2h, n, 8);
  }

  // ---- layer 2 GEMM: out = agg2h@W2l^T + hh@W2r^T + b2 ----
  {
    int waves = (NNODES / 16) * (OUTF / 64);             // 3125
    sage_gemm_f16<0><<<(waves * 32 + T - 1) / T, T, 0, stream>>>(
        agg2h, w2lh, hh, w2rh, b2, out, (_Float16*)nullptr, NNODES, OUTF, HIDDEN);
  }
}
